// Attention_19868518711699
// MI455X (gfx1250) — matrix operand order repaired and checked by reading, not compiled, not run on hardware
//
#include <hip/hip_runtime.h>

typedef __attribute__((ext_vector_type(16))) __bf16 v16bf;
typedef __attribute__((ext_vector_type(8)))  float  v8f;

union BF16x16 { v16bf v; unsigned u[8]; };

__device__ __forceinline__ unsigned short f2bf(float f) {
  union { float f; unsigned u; } x; x.f = f;
  unsigned u = x.u;
  return (unsigned short)((u + 0x7FFFu + ((u >> 16) & 1u)) >> 16);
}
__device__ __forceinline__ float bf2f(unsigned short h) {
  union { unsigned u; float f; } x; x.u = ((unsigned)h) << 16; return x.f;
}
__device__ __forceinline__ unsigned pk2(float a, float b) {
  return (unsigned)f2bf(a) | ((unsigned)f2bf(b) << 16);
}
// 16-bit A-operand K index for VGPR r, lane-half (contiguous order: k = 16*half + 2*r)
__device__ __forceinline__ int a_kk(int r, int half) {
  return 2 * r + 16 * half;
}
__device__ __forceinline__ void load_b16(BF16x16& d, const unsigned short* p) {
  uint4 a = *(const uint4*)p;
  uint4 b = *(const uint4*)(p + 8);
  d.u[0] = a.x; d.u[1] = a.y; d.u[2] = a.z; d.u[3] = a.w;
  d.u[4] = b.x; d.u[5] = b.y; d.u[6] = b.z; d.u[7] = b.w;
}
__device__ __forceinline__ v8f wmma_bf16(const BF16x16& a, const BF16x16& b, v8f c) {
  return __builtin_amdgcn_wmma_f32_16x16x32_bf16(false, a.v, false, b.v, (short)0, c, false, false);
}

#define NTOK  577
#define NPAD  608
#define HEADS 12
#define HD    64
#define CDIM  768
#define BATCH 32
#define MROWS (BATCH * NTOK)       // 18464
#define MTILES (MROWS / 16)        // 1154
#define MBLK  ((MTILES + 7) / 8)   // 145 blocks along M
#define SCALE 0.125f               // 64^-0.5

// ---------------- weight conversion ----------------
__global__ __launch_bounds__(256) void cvt_f32_bf16(const float* __restrict__ s,
                                                    unsigned short* __restrict__ d, int n) {
  int i = blockIdx.x * 256 + threadIdx.x;
  if (i < n) d[i] = f2bf(s[i]);
}

// ---------------- GEMM: out[M,O] = A[M,K] @ W[O,K]^T (+bias) ----------------
// block = 8 waves sharing one 64-wide O strip (blockIdx.y); per K-step the
// 64x32 bf16 B slice is staged in LDS once and read 8x via ds_load_b128.
template <bool ABF16, bool OUTF32>
__global__ __launch_bounds__(256) void gemm_wmma(const void* __restrict__ Ap,
                                                 const unsigned short* __restrict__ W,
                                                 const float* __restrict__ bias,
                                                 void* __restrict__ Out,
                                                 int K, int O) {
  __shared__ unsigned short Bs[64 * 32];          // 4 KB B slice
  const int tid = threadIdx.x;
  const int lane = tid & 31, wid = tid >> 5;
  const int os = blockIdx.y;                      // 64-wide O strip
  const int mt0 = blockIdx.x * 8 + wid;
  const bool active = (mt0 < MTILES);
  const int mt = active ? mt0 : (MTILES - 1);     // clamp ragged waves (store masked)
  const int m = lane & 15, half = lane >> 4;

  v8f z = {0.f, 0.f, 0.f, 0.f, 0.f, 0.f, 0.f, 0.f};
  v8f acc[4] = {z, z, z, z};

  const size_t arow = (size_t)(mt * 16 + m) * K;
  // cooperative staging: thread t copies (row = t>>2, 16-byte segment = t&3)
  const int srow = tid >> 2, sseg = tid & 3;
  const unsigned short* wsrc = W + (size_t)(os * 64 + srow) * K + sseg * 8;

  for (int c0 = 0; c0 < K; c0 += 32) {
    __syncthreads();                              // previous slice fully consumed
    *(uint4*)&Bs[srow * 32 + sseg * 8] = *(const uint4*)(wsrc + c0);
    if (c0 + 32 < K) __builtin_prefetch(wsrc + c0 + 32, 0, 1);
    __syncthreads();

    BF16x16 a;
    if constexpr (ABF16) {
      const unsigned short* A = (const unsigned short*)Ap;
#pragma unroll
      for (int r = 0; r < 8; ++r)
        a.u[r] = *(const unsigned*)(A + arow + c0 + a_kk(r, half));
    } else {
      const float* A = (const float*)Ap;
#pragma unroll
      for (int r = 0; r < 8; ++r) {
        float2 f = *(const float2*)(A + arow + c0 + a_kk(r, half));
        a.u[r] = pk2(f.x, f.y);
      }
    }
#pragma unroll
    for (int g = 0; g < 4; ++g) {
      const int base = (g * 16 + m) * 32 + 16 * half;   // B lane = N column
      BF16x16 b;
      uint4 t0 = *(const uint4*)&Bs[base];
      uint4 t1 = *(const uint4*)&Bs[base + 8];
      b.u[0] = t0.x; b.u[1] = t0.y; b.u[2] = t0.z; b.u[3] = t0.w;
      b.u[4] = t1.x; b.u[5] = t1.y; b.u[6] = t1.z; b.u[7] = t1.w;
      acc[g] = wmma_bf16(a, b, acc[g]);
    }
  }
  if (!active) return;
#pragma unroll
  for (int g = 0; g < 4; ++g)
#pragma unroll
    for (int r = 0; r < 8; ++r) {
      const int row = mt * 16 + r + 8 * half;     // C layout: M = r + 8*half
      const int col = os * 64 + g * 16 + m;
      float v = acc[g][r];
      if constexpr (OUTF32) {
        v += bias[col];
        ((float*)Out)[(size_t)row * O + col] = v;
      } else {
        ((unsigned short*)Out)[(size_t)row * O + col] = f2bf(v);
      }
    }
}

// ---------------- RoPE + pack q,k row-major / v transposed ----------------
__global__ __launch_bounds__(256) void rope_pack(const unsigned short* __restrict__ qkv,
                                                 const float* __restrict__ cosb,
                                                 const float* __restrict__ sinb,
                                                 unsigned short* __restrict__ qp,
                                                 unsigned short* __restrict__ kp,
                                                 unsigned short* __restrict__ vT) {
  int i = blockIdx.x * 256 + threadIdx.x;
  const int TOT = BATCH * HEADS * NPAD;
  if (i >= TOT) return;
  const int nn = i % NPAD;
  const int bh = i / NPAD;
  const int h = bh % HEADS, b = bh / HEADS;
  unsigned short* qrow = qp + ((size_t)bh * NPAD + nn) * HD;
  unsigned short* krow = kp + ((size_t)bh * NPAD + nn) * HD;
  if (nn >= NTOK) {  // zero padding rows / columns
#pragma unroll
    for (int d = 0; d < HD; ++d) {
      qrow[d] = 0; krow[d] = 0;
      vT[((size_t)bh * HD + d) * NPAD + nn] = 0;
    }
    return;
  }
  const size_t tb = (size_t)(b * NTOK + nn) * (3 * CDIM);
  const unsigned short* qs = qkv + tb + (size_t)h * HD;
  const unsigned short* ks = qkv + tb + (size_t)(HEADS + h) * HD;
  const unsigned short* vs = qkv + tb + (size_t)(2 * HEADS + h) * HD;
#pragma unroll
  for (int j = 0; j < 32; ++j) {
    float q1 = bf2f(qs[2 * j]), q2 = bf2f(qs[2 * j + 1]);
    float k1 = bf2f(ks[2 * j]), k2 = bf2f(ks[2 * j + 1]);
    float c = 1.f, s = 0.f;
    if (nn > 0) { c = cosb[(nn - 1) * 32 + j]; s = sinb[(nn - 1) * 32 + j]; }
    qrow[2 * j]     = f2bf(q1 * c - q2 * s);
    qrow[2 * j + 1] = f2bf(q1 * s + q2 * c);
    krow[2 * j]     = f2bf(k1 * c - k2 * s);
    krow[2 * j + 1] = f2bf(k1 * s + k2 * c);
  }
#pragma unroll
  for (int d = 0; d < HD; ++d)
    vT[((size_t)bh * HD + d) * NPAD + nn] = vs[d];
}

// ---------------- flash attention, one wave per 16-token q tile ----------------
__global__ __launch_bounds__(256) void attn_fa(const unsigned short* __restrict__ qp,
                                               const unsigned short* __restrict__ kp,
                                               const unsigned short* __restrict__ vT,
                                               unsigned short* __restrict__ attnb) {
  __shared__ unsigned short Ps[8][16 * 32];   // per-wave C-layout -> A-layout bounce tile
  const int lane = threadIdx.x & 31, wid = threadIdx.x >> 5;
  const int bh = blockIdx.x;
  const int b = bh / HEADS, h = bh % HEADS;
  const int qt = blockIdx.y * 8 + wid;
  if (qt >= 37) return;
  const int m = lane & 15, half = lane >> 4;

  BF16x16 qa[2];
  {
    const size_t qb = ((size_t)bh * NPAD + qt * 16 + m) * HD;
#pragma unroll
    for (int dh = 0; dh < 2; ++dh)
#pragma unroll
      for (int r = 0; r < 8; ++r)
        qa[dh].u[r] = *(const unsigned*)(qp + qb + 32 * dh + a_kk(r, half));
  }

  v8f z = {0.f, 0.f, 0.f, 0.f, 0.f, 0.f, 0.f, 0.f};
  v8f acc[4] = {z, z, z, z};
  float mrow[8], lrow[8];
#pragma unroll
  for (int r = 0; r < 8; ++r) { mrow[r] = -1e30f; lrow[r] = 0.f; }

  const unsigned short* kbase = kp + (size_t)bh * NPAD * HD;
  const unsigned short* vbase = vT + (size_t)bh * HD * NPAD;

  for (int kb = 0; kb < NPAD / 32; ++kb) {
    if (kb + 1 < NPAD / 32) {   // cover latency of the next L2-resident chunk
      __builtin_prefetch(kbase + (size_t)((kb + 1) * 32 + m) * HD + 16 * half, 0, 1);
      __builtin_prefetch(kbase + (size_t)((kb + 1) * 32 + 16 + m) * HD + 16 * half, 0, 1);
      __builtin_prefetch(vbase + (size_t)m * NPAD + (kb + 1) * 32, 0, 1);
    }
    // S = Q @ K^T for 32 keys (two 16-wide subtiles)
    v8f S0 = z, S1 = z;
    {
      const unsigned short* kr = kbase + (size_t)(kb * 32 + m) * HD + 16 * half;
      BF16x16 b0, b1; load_b16(b0, kr); load_b16(b1, kr + 32);
      S0 = wmma_bf16(qa[0], b0, S0);
      S0 = wmma_bf16(qa[1], b1, S0);
    }
    {
      const unsigned short* kr = kbase + (size_t)(kb * 32 + 16 + m) * HD + 16 * half;
      BF16x16 b0, b1; load_b16(b0, kr); load_b16(b1, kr + 32);
      S1 = wmma_bf16(qa[0], b0, S1);
      S1 = wmma_bf16(qa[1], b1, S1);
    }
    const bool mask0 = (kb * 32 + m) >= NTOK;
    const bool mask1 = (kb * 32 + 16 + m) >= NTOK;
    // online softmax per row (row = r + 8*half; 16-lane group = the 16 columns)
#pragma unroll
    for (int r = 0; r < 8; ++r) {
      float s0 = mask0 ? -1e30f : S0[r] * SCALE;
      float s1 = mask1 ? -1e30f : S1[r] * SCALE;
      float lm = fmaxf(s0, s1);
      lm = fmaxf(lm, __shfl_xor(lm, 1));
      lm = fmaxf(lm, __shfl_xor(lm, 2));
      lm = fmaxf(lm, __shfl_xor(lm, 4));
      lm = fmaxf(lm, __shfl_xor(lm, 8));
      float mnew = fmaxf(mrow[r], lm);
      float alpha = __expf(mrow[r] - mnew);
      mrow[r] = mnew;
      float p0 = __expf(s0 - mnew);
      float p1 = __expf(s1 - mnew);
      float rs = p0 + p1;
      rs += __shfl_xor(rs, 1);
      rs += __shfl_xor(rs, 2);
      rs += __shfl_xor(rs, 4);
      rs += __shfl_xor(rs, 8);
      lrow[r] = lrow[r] * alpha + rs;
#pragma unroll
      for (int g = 0; g < 4; ++g) acc[g][r] *= alpha;
      const int mr = r + 8 * half;
      Ps[wid][mr * 32 + m]      = f2bf(p0);
      Ps[wid][mr * 32 + 16 + m] = f2bf(p1);
    }
    // wave-local store->load ordering on LDS (CDNA5 split counter)
    asm volatile("s_wait_dscnt 0" ::: "memory");
    // P back out in A-operand layout
    BF16x16 pa;
#pragma unroll
    for (int r = 0; r < 8; ++r)
      pa.u[r] = *(const unsigned*)&Ps[wid][m * 32 + a_kk(r, half)];
    // acc += P @ V  (V^T gives contiguous key runs per d)
#pragma unroll
    for (int g = 0; g < 4; ++g) {
      const int d = g * 16 + m;
      BF16x16 vb;
      load_b16(vb, vbase + (size_t)d * NPAD + kb * 32 + 16 * half);
      acc[g] = wmma_bf16(pa, vb, acc[g]);
    }
  }
#pragma unroll
  for (int r = 0; r < 8; ++r) lrow[r] = 1.f / lrow[r];
#pragma unroll
  for (int g = 0; g < 4; ++g)
#pragma unroll
    for (int r = 0; r < 8; ++r) {
      const int t = qt * 16 + r + 8 * half;
      if (t < NTOK) {
        float o = acc[g][r] * lrow[r];
        attnb[((size_t)b * NTOK + t) * CDIM + h * HD + g * 16 + m] = f2bf(o);
      }
    }
}

// ---------------- host-side orchestration ----------------
extern "C" void kernel_launch(void* const* d_in, const int* in_sizes, int n_in,
                              void* d_out, int out_size, void* d_ws, size_t ws_size,
                              hipStream_t stream) {
  const float* x      = (const float*)d_in[0];   // [32,577,768]
  const float* w_qkv  = (const float*)d_in[1];   // [2304,768]
  const float* w_proj = (const float*)d_in[2];   // [768,768]
  const float* b_proj = (const float*)d_in[3];   // [768]
  const float* cosb   = (const float*)d_in[4];   // [576,32]
  const float* sinb   = (const float*)d_in[5];   // [576,32]
  float* out = (float*)d_out;

  char* ws = (char*)d_ws;
  const size_t SZ_WQKV = (size_t)3 * CDIM * CDIM * 2;          // 3.5 MB bf16
  const size_t SZ_WPRJ = (size_t)CDIM * CDIM * 2;              // 1.2 MB bf16
  const size_t SZ_QKV  = (size_t)MROWS * 3 * CDIM * 2;         // 85 MB bf16
  const size_t SZ_PACK = (size_t)BATCH * HEADS * NPAD * HD * 2;// 30 MB each

  unsigned short* wqkvb = (unsigned short*)(ws);
  unsigned short* wprjb = (unsigned short*)(ws + SZ_WQKV);
  unsigned short* qkvb  = (unsigned short*)(ws + SZ_WQKV + SZ_WPRJ);
  unsigned short* qp    = (unsigned short*)(ws + SZ_WQKV + SZ_WPRJ + SZ_QKV);
  unsigned short* kp    = (unsigned short*)((char*)qp + SZ_PACK);
  unsigned short* vT    = (unsigned short*)((char*)kp + SZ_PACK);
  unsigned short* attnb = qkvb;  // qkv dead after rope_pack; reuse (28 MB <= 85 MB)

  // 1-2) weights -> bf16
  {
    int n1 = 3 * CDIM * CDIM, n2 = CDIM * CDIM;
    cvt_f32_bf16<<<(n1 + 255) / 256, 256, 0, stream>>>(w_qkv, wqkvb, n1);
    cvt_f32_bf16<<<(n2 + 255) / 256, 256, 0, stream>>>(w_proj, wprjb, n2);
  }
  // 3) qkv = x @ w_qkv^T   (M=18464, K=768, O=2304) -> bf16
  gemm_wmma<false, false><<<dim3(MBLK, 3 * CDIM / 64), 256, 0, stream>>>(
      x, wqkvb, nullptr, qkvb, CDIM, 3 * CDIM);
  // 4) RoPE + pack
  {
    int tot = BATCH * HEADS * NPAD;
    rope_pack<<<(tot + 255) / 256, 256, 0, stream>>>(qkvb, cosb, sinb, qp, kp, vT);
  }
  // 5) flash attention -> attnb [B,N,C] bf16
  attn_fa<<<dim3(BATCH * HEADS, 5), 256, 0, stream>>>(qp, kp, vT, attnb);
  // 6) out = attn @ w_proj^T + b_proj   (M=18464, K=768, O=768) -> f32
  gemm_wmma<true, true><<<dim3(MBLK, CDIM / 64), 256, 0, stream>>>(
      attnb, wprjb, b_proj, out, CDIM, CDIM);
}
